// LiGRU_Layer_19842748907936
// MI455X (gfx1250) — matrix operand order repaired and checked by reading, not compiled, not run on hardware
//
#include <hip/hip_runtime.h>
#include <math.h>

// ---------------- types ----------------
typedef _Float16 f16;
typedef __attribute__((ext_vector_type(16))) _Float16 v16h;
typedef __attribute__((ext_vector_type(8)))  _Float16 v8h;
typedef __attribute__((ext_vector_type(8)))  float    v8f;

// problem constants
#define BB   16
#define TT   2000
#define DD   512
#define HH   512
#define GG   1024           // 2*H
#define NWG  32             // persistent workgroups in recurrence

// workspace layout (bytes)
#define WS_BAR    0                       // 256 B barrier
#define WS_HBUF   256                     // 2 x (16*512) f16 ping-pong = 32768
#define WS_XH     33024                   // x in f16: 16*2000*512*2 = 32768000
#define WS_WH     32801024                // W in f16: 1024*512*2 = 1048576
#define WS_WBN    33849600                // wbn (T,B,G) f16: 2000*16*1024*2 = 65536000
// total ~99.4 MB

// ---------------- helpers ----------------
__device__ __forceinline__ v16h make_frag_a(const f16* base, int kb) {
    // lane holds row (lane&15); elems j = K kb+j (16 contiguous)
    v8h lo = *(const v8h*)(base + kb);
    v8h hi = *(const v8h*)(base + kb + 8);
    v16h a;
#pragma unroll
    for (int e = 0; e < 8; ++e) { a[e] = lo[e]; a[8 + e] = hi[e]; }
    return a;
}
__device__ __forceinline__ v16h make_frag_b(const f16* colbase, int ko) {
    // lane holds col (lane&15); elems j = K ko+j (16 contiguous)
    v8h lo = *(const v8h*)(colbase + ko);
    v8h hi = *(const v8h*)(colbase + ko + 8);
    v16h b;
#pragma unroll
    for (int e = 0; e < 8; ++e) { b[e] = lo[e]; b[8 + e] = hi[e]; }
    return b;
}

// ---------------- init ----------------
__global__ void init_bar_kernel(unsigned* bar) {
    if (threadIdx.x < 64) bar[threadIdx.x] = 0u;
}

__global__ void cvt_f32_to_f16_kernel(const float* __restrict__ src,
                                      f16* __restrict__ dst, int n) {
    int i = blockIdx.x * blockDim.x + threadIdx.x;
    int stride = gridDim.x * blockDim.x;
    for (; i < n; i += stride) dst[i] = (f16)src[i];
}

// ---------------- phase 1: wbn = BN(x @ W^T), stored (T,B,G) f16 ----------------
// grid (2000, 2), block 256 (8 waves). wave -> 64-col strip (4 WMMA tiles).
__global__ __launch_bounds__(256) void gemm_bn_kernel(
        const f16* __restrict__ xh,     // (B*T, 512)
        const f16* __restrict__ Wh,     // (1024, 512)
        const float* __restrict__ gamma, const float* __restrict__ beta,
        const float* __restrict__ rmean, const float* __restrict__ rvar,
        f16* __restrict__ wbn)          // (T, 16, 1024)
{
    const int lane = threadIdx.x & 31;
    const int wave = threadIdx.x >> 5;
    const int mtile = blockIdx.x;                       // 0..1999
    const int gbase = blockIdx.y * 512 + wave * 64;

    const int row = lane & 15;
    const int hi  = lane >> 4;
    const f16* aptr = xh + (size_t)(mtile * 16 + row) * DD;

    v8f acc[4] = {};
    for (int kt = 0; kt < 16; ++kt) {
        v16h afrag = make_frag_a(aptr, kt * 32 + hi * 16);
        int ko = kt * 32 + hi * 16;
#pragma unroll
        for (int j = 0; j < 4; ++j) {
            const f16* bptr = Wh + (size_t)(gbase + j * 16 + row) * DD;
            v16h bfrag = make_frag_b(bptr, ko);
            acc[j] = __builtin_amdgcn_wmma_f32_16x16x32_f16(
                false, afrag, false, bfrag, (short)0, acc[j], false, false);
        }
    }
#pragma unroll
    for (int j = 0; j < 4; ++j) {
        int g = gbase + j * 16 + row;                   // this lane's column
        float inv = gamma[g] * rsqrtf(rvar[g] + 1e-5f);
        float bb  = beta[g] - rmean[g] * inv;
#pragma unroll
        for (int r = 0; r < 8; ++r) {
            int m = mtile * 16 + hi * 8 + r;            // m = b*T + t
            int b = m / TT;
            int t = m - b * TT;
            wbn[(size_t)(t * 16 + b) * GG + g] = (f16)(acc[j][r] * inv + bb);
        }
    }
}

// ---------------- phase 2: persistent recurrence ----------------
__device__ __forceinline__ void grid_sync(unsigned* bar, unsigned gen) {
    __threadfence();
    __syncthreads();
    if (threadIdx.x == 0) {
        atomicAdd(bar, 1u);
        while (__hip_atomic_load(bar, __ATOMIC_RELAXED, __HIP_MEMORY_SCOPE_AGENT)
               < gen * (unsigned)NWG) {
            __builtin_amdgcn_s_sleep(2);
        }
    }
    __syncthreads();
    __threadfence();
}

// grid 32 WGs x 64 threads (2 waves). WG i owns h columns [i*16, i*16+16).
// wave0 -> 'a' gate tile (cols i*16..), wave1 -> 'z' gate tile (cols 512+i*16..).
__global__ __launch_bounds__(64) void ligru_rec_kernel(
        const f16* __restrict__ wbn,    // (T,16,1024) f16
        const float* __restrict__ U,    // (1024,512) f32
        float* __restrict__ out,        // (16,2000,512) f32
        f16* __restrict__ hbuf,         // 2 x (16,512) f16 ping-pong
        unsigned* __restrict__ bar)
{
    __shared__ f16   Us[32 * 512];      // this WG's U slice, B-friendly (32 KB)
    __shared__ float gates[32 * 16];    // [local col][b]
    __shared__ float hloc[16 * 16];     // [b][j] own h columns, f32 exact

    const int wg   = blockIdx.x;
    const int tid  = threadIdx.x;
    const int lane = tid & 31;
    const int wave = tid >> 5;

    // prologue: cache U slice into LDS as f16 (rows = our 32 gate cols)
    for (int idx = tid; idx < 32 * 512; idx += 64) {
        int c = idx >> 9;
        int k = idx & 511;
        int g = (c < 16) ? (wg * 16 + c) : (512 + wg * 16 + (c - 16));
        Us[idx] = (f16)U[(size_t)g * DD + k];
    }
    for (int idx = tid; idx < 256; idx += 64) hloc[idx] = 0.f;
    for (int idx = tid; idx < 256; idx += 64) {         // zero h ping buffer 0 (our slice)
        int b = idx >> 4, j = idx & 15;
        hbuf[b * HH + wg * 16 + j] = (f16)0.f;
    }
    grid_sync(bar, 1u);                                  // all slices of h0 visible

    const int row = lane & 15;                           // batch row / N col
    const int hi  = lane >> 4;
    const int cb  = wave * 16;                           // local col base (0 = a, 16 = z)
    const f16* bcol = Us + (size_t)(cb + row) * DD;
    const int gcol = (wave == 0) ? (wg * 16 + row) : (512 + wg * 16 + row);

    for (int t = 0; t < TT; ++t) {
        const f16* hread  = hbuf + (size_t)(t & 1) * (16 * HH);
        f16*       hwrite = hbuf + (size_t)((t + 1) & 1) * (16 * HH);

        // prefetch w_t+4 slab (speculative, advisory)
        if (t + 4 < TT) {
            const char* p = (const char*)(wbn + (size_t)(t + 4) * 16 * GG) + tid * 512;
            __builtin_prefetch(p, 0, 1);
        }

        v8f acc = {};
        const f16* aptr = hread + (size_t)row * HH;      // A row = batch b
#pragma unroll
        for (int kt = 0; kt < 16; ++kt) {
            v16h afrag = make_frag_a(aptr, kt * 32 + hi * 16);
            v16h bfrag = make_frag_b(bcol, kt * 32 + hi * 16);
            acc = __builtin_amdgcn_wmma_f32_16x16x32_f16(
                false, afrag, false, bfrag, (short)0, acc, false, false);
        }
        // add w_t and park gate tile in LDS
#pragma unroll
        for (int r = 0; r < 8; ++r) {
            int b = hi * 8 + r;
            float wv = (float)wbn[(size_t)(t * 16 + b) * GG + gcol];
            gates[(cb + row) * 16 + b] = acc[r] + wv;
        }
        __syncthreads();

        // gating update for our 16 h columns (256 elements)
        for (int idx = tid; idx < 256; idx += 64) {
            int b = idx >> 4, j = idx & 15;
            float av = gates[j * 16 + b];                // 'a' tile
            float zv = gates[(16 + j) * 16 + b];         // 'z' tile
            float z  = 1.f / (1.f + __expf(-zv));
            float hv = hloc[idx];
            float hc = av > 0.f ? av : 0.f;
            float hn = z * hv + (1.f - z) * hc;
            hloc[idx] = hn;
            out[((size_t)b * TT + t) * HH + wg * 16 + j] = hn;
            hwrite[b * HH + wg * 16 + j] = (f16)hn;
        }
        grid_sync(bar, (unsigned)(t + 2));               // publish h_t, fetch-fence
    }
}

// ---------------- launch ----------------
extern "C" void kernel_launch(void* const* d_in, const int* in_sizes, int n_in,
                              void* d_out, int out_size, void* d_ws, size_t ws_size,
                              hipStream_t stream) {
    const float* x     = (const float*)d_in[0];
    const float* W     = (const float*)d_in[1];
    const float* U     = (const float*)d_in[2];
    const float* gamma = (const float*)d_in[3];
    const float* beta  = (const float*)d_in[4];
    const float* rmean = (const float*)d_in[5];
    const float* rvar  = (const float*)d_in[6];
    float* out = (float*)d_out;

    char* ws = (char*)d_ws;
    unsigned* bar = (unsigned*)(ws + WS_BAR);
    f16* hbuf = (f16*)(ws + WS_HBUF);
    f16* xh   = (f16*)(ws + WS_XH);
    f16* Wh   = (f16*)(ws + WS_WH);
    f16* wbn  = (f16*)(ws + WS_WBN);

    init_bar_kernel<<<1, 64, 0, stream>>>(bar);

    const int nx = BB * TT * DD;
    cvt_f32_to_f16_kernel<<<2048, 256, 0, stream>>>(x, xh, nx);
    cvt_f32_to_f16_kernel<<<512, 256, 0, stream>>>(W, Wh, GG * DD);

    gemm_bn_kernel<<<dim3(TT, 2), 256, 0, stream>>>(xh, Wh, gamma, beta,
                                                    rmean, rvar, wbn);

    ligru_rec_kernel<<<NWG, 64, 0, stream>>>(wbn, U, out, hbuf, bar);
}
